// CausalSelfAttention_54468775247958
// MI455X (gfx1250) — matrix operand order repaired and checked by reading, not compiled, not run on hardware
//
#include <hip/hip_runtime.h>

// CDNA5 / gfx1250, wave32. Matrix math via v_wmma_f32_16x16x32_bf16;
// global->LDS tile staging via GLOBAL_LOAD_ASYNC_TO_LDS_B128 (ASYNCcnt DMA).
// B=4, T=2048, C=1024, H=16, D=64.

typedef __attribute__((ext_vector_type(16))) __bf16 v16bf;
typedef __attribute__((ext_vector_type(8)))  float  v8f;
typedef __attribute__((__vector_size__(16))) int    v4i;

#if defined(__AMDGCN__) && __has_builtin(__builtin_amdgcn_global_load_async_to_lds_b128)
#define ASYNC_LDS 1
typedef __attribute__((address_space(1))) v4i* gv4i_p;  // global (device) ptr
typedef __attribute__((address_space(3))) v4i* lv4i_p;  // LDS (shared) ptr
#else
#define ASYNC_LDS 0
#endif

union Frag {
    v16bf v;
    uint4 q[2];
};

__device__ __forceinline__ v8f wmma_bf16(const Frag& a, const Frag& b, v8f c) {
    return __builtin_amdgcn_wmma_f32_16x16x32_bf16(
        false, a.v, false, b.v, (short)0, c, false, false);
}

// 16-byte global -> LDS copy. Async DMA path on gfx1250 (no VGPR round trip).
__device__ __forceinline__ void async_copy16(const void* g, void* l) {
#if ASYNC_LDS
    __builtin_amdgcn_global_load_async_to_lds_b128(
        (gv4i_p)(void*)g, (lv4i_p)(void*)l, 0, 0);
#else
    *(uint4*)l = *(const uint4*)g;
#endif
}

__device__ __forceinline__ void wait_async() {
#if ASYNC_LDS
#if __has_builtin(__builtin_amdgcn_s_wait_asynccnt)
    __builtin_amdgcn_s_wait_asynccnt(0);
#else
    asm volatile("s_wait_asynccnt 0x0" ::: "memory");
#endif
#endif
}

// ---------------------------------------------------------------------------
// fp32 -> bf16 elementwise conversion (vectorized float4 -> 4x bf16).
// ---------------------------------------------------------------------------
__global__ void __launch_bounds__(256)
f32_to_bf16(const float* __restrict__ src, __bf16* __restrict__ dst, int n4) {
    int i = blockIdx.x * 256 + threadIdx.x;
    if (i < n4) {
        float4 d = ((const float4*)src)[i];
        union { __bf16 h[4]; uint2 u; } t;
        t.h[0] = (__bf16)d.x; t.h[1] = (__bf16)d.y;
        t.h[2] = (__bf16)d.z; t.h[3] = (__bf16)d.w;
        *(uint2*)(dst + 4 * (size_t)i) = t.u;
    }
}

// ---------------------------------------------------------------------------
// GEMM: C[M,N] = A[M,K](bf16) * W[N,K](bf16)^T + bias.
// 128x128x64 block tile, 8 waves, double-buffered async-LDS staging.
// MODE 0: epilogue scatters q/k/v^T bf16 for attention.
// MODE 1: epilogue stores fp32 + bias.
// ---------------------------------------------------------------------------
template <int MODE>
__global__ void __launch_bounds__(256)
gemm_wmma(const __bf16* __restrict__ A, const __bf16* __restrict__ W,
          const float* __restrict__ bias,
          __bf16* __restrict__ qout, __bf16* __restrict__ kout,
          __bf16* __restrict__ vtout, float* __restrict__ out,
          int Ntot, int Ktot)
{
    constexpr int BK = 64;
    __shared__ __attribute__((aligned(16))) __bf16 As[2][128 * BK]; // [m][k]
    __shared__ __attribute__((aligned(16))) __bf16 Bs[2][128 * BK]; // [n][k]

    const int tid  = threadIdx.x;
    const int lane = tid & 31;
    const int wid  = tid >> 5;   // 0..7
    const int wm   = wid >> 1;   // 0..3 (32-row strip)
    const int wn   = wid & 1;    // 0..1 (64-col strip)
    const int half = lane >> 4;  // 0/1
    const int l16  = lane & 15;

    const int m0 = blockIdx.y * 128;
    const int n0 = blockIdx.x * 128;

    v8f acc[2][4];
    const v8f vz = {0.f, 0.f, 0.f, 0.f, 0.f, 0.f, 0.f, 0.f};
    for (int i = 0; i < 2; ++i)
        for (int j = 0; j < 4; ++j) acc[i][j] = vz;

    // stage A+B tiles for k-step into LDS buffer `buf` (async DMA, 16B/lane x4)
    auto stage = [&](int buf, int k0) {
        #pragma unroll
        for (int i = 0; i < 4; ++i) {
            int f = tid + i * 256;
            int r = f >> 3;            // 8 x uint4 per 64-elem row
            int c = (f & 7) << 3;
            async_copy16(A + (size_t)(m0 + r) * Ktot + k0 + c, &As[buf][r * BK + c]);
            async_copy16(W + (size_t)(n0 + r) * Ktot + k0 + c, &Bs[buf][r * BK + c]);
        }
    };

    const int nk = Ktot / BK;
    stage(0, 0);
    wait_async();
    __syncthreads();

    for (int ks = 0; ks < nk; ++ks) {
        const int buf = ks & 1;
        if (ks + 1 < nk) stage(buf ^ 1, (ks + 1) * BK); // DMA next tile during compute

        const __bf16* Ab = As[buf];
        const __bf16* Bb = Bs[buf];
        #pragma unroll
        for (int kc = 0; kc < BK; kc += 32) {
            Frag a[2], b[4];
            #pragma unroll
            for (int tm = 0; tm < 2; ++tm) {
                const __bf16* ap = &Ab[(wm * 32 + tm * 16 + l16) * BK + kc];
                a[tm].q[0] = *(const uint4*)(ap + half * 8);        // K 0..7 / 8..15
                a[tm].q[1] = *(const uint4*)(ap + 16 + half * 8);   // K 16..23 / 24..31
            }
            #pragma unroll
            for (int tn = 0; tn < 4; ++tn) {
                const __bf16* bp = &Bb[(wn * 64 + tn * 16 + l16) * BK + kc];
                b[tn].q[0] = *(const uint4*)(bp + half * 8);
                b[tn].q[1] = *(const uint4*)(bp + 16 + half * 8);
            }
            #pragma unroll
            for (int tm = 0; tm < 2; ++tm)
                #pragma unroll
                for (int tn = 0; tn < 4; ++tn)
                    acc[tm][tn] = wmma_bf16(a[tm], b[tn], acc[tm][tn]);
        }
        wait_async();
        __syncthreads();
    }

    // ---- epilogue ----
    #pragma unroll
    for (int tm = 0; tm < 2; ++tm) {
        #pragma unroll
        for (int tn = 0; tn < 4; ++tn) {
            const int n  = n0 + wn * 64 + tn * 16 + l16;
            const float bv = bias[n];
            const int mbase = m0 + wm * 32 + tm * 16 + half * 8;
            #pragma unroll
            for (int e = 0; e < 8; ++e) {
                const int m = mbase + e;
                const float val = acc[tm][tn][e] + bv;
                if (MODE == 1) {
                    out[(size_t)m * Ntot + n] = val;
                } else {
                    const int sel = n >> 10;      // 0:q 1:k 2:v
                    const int c   = n & 1023;
                    const int hh  = c >> 6;
                    const int d   = c & 63;
                    const int bI  = m >> 11;
                    const int t   = m & 2047;
                    const size_t bh = (size_t)(bI * 16 + hh);
                    if (sel == 0)
                        qout[(bh * 2048 + t) * 64 + d] = (__bf16)val;
                    else if (sel == 1)
                        kout[(bh * 2048 + t) * 64 + d] = (__bf16)val;
                    else // V stored transposed [B,H,D,T] for contiguous P*V B-frags
                        vtout[(bh * 64 + d) * 2048 + t] = (__bf16)val;
                }
            }
        }
    }
}

// ---------------------------------------------------------------------------
// Flash attention: one workgroup = (b, h, 128 query rows); 8 waves x 16 rows.
// Running-softmax over 128-key blocks; S and P*V via WMMA bf16; K/V^T staged
// with async-LDS DMA.
// ---------------------------------------------------------------------------
__global__ void __launch_bounds__(256)
flash_attn(const __bf16* __restrict__ q, const __bf16* __restrict__ k,
           const __bf16* __restrict__ vt, __bf16* __restrict__ y)
{
    __shared__ __attribute__((aligned(16))) __bf16 Ks[128 * 64];       // [key][d]
    __shared__ __attribute__((aligned(16))) __bf16 Vs[64 * 128];       // [d][key]
    __shared__ __attribute__((aligned(16))) __bf16 Ps[8 * 16 * 128];   // per-wave P

    const int tid  = threadIdx.x;
    const int lane = tid & 31;
    const int w    = tid >> 5;
    const int half = lane >> 4;
    const int l16  = lane & 15;
    const int qb   = blockIdx.x * 128;
    const int h    = blockIdx.y;
    const int b    = blockIdx.z;
    const size_t bh = (size_t)(b * 16 + h);

    // Q fragments for this wave's 16 rows (registers for whole kernel)
    Frag aq[2];
    {
        const __bf16* qp = q + (bh * 2048 + qb + w * 16 + l16) * 64;
        #pragma unroll
        for (int kc = 0; kc < 2; ++kc) {
            aq[kc].q[0] = *(const uint4*)(qp + kc * 32 + half * 8);
            aq[kc].q[1] = *(const uint4*)(qp + kc * 32 + 16 + half * 8);
        }
    }

    float mrun[8], lrun[8];
    v8f o[4];
    const v8f vz = {0.f, 0.f, 0.f, 0.f, 0.f, 0.f, 0.f, 0.f};
    #pragma unroll
    for (int i = 0; i < 4; ++i) o[i] = vz;
    #pragma unroll
    for (int e = 0; e < 8; ++e) { mrun[e] = -1e30f; lrun[e] = 0.f; }

    for (int kb = 0; kb <= qb; kb += 128) {
        // ---- async-stage K block [128][64] and V^T block [64][128] ----
        #pragma unroll
        for (int i = 0; i < 4; ++i) {
            int f = tid + i * 256;
            int r = f >> 3, c = (f & 7) << 3;
            async_copy16(k + (bh * 2048 + kb + r) * 64 + c, &Ks[r * 64 + c]);
            int r2 = f >> 4, c2 = (f & 15) << 3;
            async_copy16(vt + (bh * 64 + r2) * 2048 + kb + c2, &Vs[r2 * 128 + c2]);
        }
        wait_async();
        __syncthreads();

        // ---- S = Q K^T : 8 key tiles x 2 K-chunks ----
        v8f s[8];
        #pragma unroll
        for (int tn = 0; tn < 8; ++tn) s[tn] = vz;
        #pragma unroll
        for (int kc = 0; kc < 2; ++kc) {
            #pragma unroll
            for (int tn = 0; tn < 8; ++tn) {
                Frag bk;
                const __bf16* kp = &Ks[(tn * 16 + l16) * 64 + kc * 32];
                bk.q[0] = *(const uint4*)(kp + half * 8);
                bk.q[1] = *(const uint4*)(kp + 16 + half * 8);
                s[tn] = wmma_bf16(aq[kc], bk, s[tn]);
            }
        }

        // ---- scale + causal mask (diagonal block only) ----
        const bool diag = (kb == qb);
        #pragma unroll
        for (int tn = 0; tn < 8; ++tn)
            #pragma unroll
            for (int e = 0; e < 8; ++e) {
                float sv = s[tn][e] * 0.125f;   // 1/sqrt(64)
                if (diag) {
                    int qi = w * 16 + half * 8 + e;
                    int kj = tn * 16 + l16;
                    if (kj > qi) sv = -1e30f;
                }
                s[tn][e] = sv;
            }

        // ---- running softmax stats (reduce across 16-lane half-groups) ----
        float mnew[8], alpha[8];
        #pragma unroll
        for (int e = 0; e < 8; ++e) {
            float mx = s[0][e];
            #pragma unroll
            for (int tn = 1; tn < 8; ++tn) mx = fmaxf(mx, s[tn][e]);
            #pragma unroll
            for (int off = 1; off < 16; off <<= 1)
                mx = fmaxf(mx, __shfl_xor(mx, off, 32));
            mnew[e]  = fmaxf(mrun[e], mx);
            alpha[e] = __expf(mrun[e] - mnew[e]);
            mrun[e]  = mnew[e];
        }
        #pragma unroll
        for (int e = 0; e < 8; ++e) {
            float rs = 0.f;
            #pragma unroll
            for (int tn = 0; tn < 8; ++tn) {
                float p = __expf(s[tn][e] - mnew[e]);
                s[tn][e] = p;
                rs += p;
            }
            #pragma unroll
            for (int off = 1; off < 16; off <<= 1)
                rs += __shfl_xor(rs, off, 32);
            lrun[e] = lrun[e] * alpha[e] + rs;
        }
        #pragma unroll
        for (int td = 0; td < 4; ++td)
            #pragma unroll
            for (int e = 0; e < 8; ++e) o[td][e] *= alpha[e];

        // ---- P -> LDS (wave-private region), C-layout -> A-layout ----
        __bf16* pw = &Ps[w * 16 * 128];
        #pragma unroll
        for (int tn = 0; tn < 8; ++tn) {
            int n = tn * 16 + l16;
            #pragma unroll
            for (int e = 0; e < 8; ++e)
                pw[(half * 8 + e) * 128 + n] = (__bf16)s[tn][e];
        }

        // ---- O += P V : 4 key chunks x 4 d tiles ----
        #pragma unroll
        for (int kc = 0; kc < 4; ++kc) {
            Frag ap;
            const __bf16* pp = &pw[l16 * 128 + kc * 32];
            ap.q[0] = *(const uint4*)(pp + half * 8);
            ap.q[1] = *(const uint4*)(pp + 16 + half * 8);
            #pragma unroll
            for (int td = 0; td < 4; ++td) {
                Frag bv;
                const __bf16* vp = &Vs[(td * 16 + l16) * 128 + kc * 32];
                bv.q[0] = *(const uint4*)(vp + half * 8);
                bv.q[1] = *(const uint4*)(vp + 16 + half * 8);
                o[td] = wmma_bf16(ap, bv, o[td]);
            }
        }
        __syncthreads();
    }

    // ---- normalize and store y_att as bf16 [B,T,C] ----
    #pragma unroll
    for (int td = 0; td < 4; ++td) {
        #pragma unroll
        for (int e = 0; e < 8; ++e) {
            float val = o[td][e] / lrun[e];
            int t = qb + w * 16 + half * 8 + e;
            int d = td * 16 + l16;
            y[((size_t)b * 2048 + t) * 1024 + h * 64 + d] = (__bf16)val;
        }
    }
}

// ---------------------------------------------------------------------------
extern "C" void kernel_launch(void* const* d_in, const int* in_sizes, int n_in,
                              void* d_out, int out_size, void* d_ws, size_t ws_size,
                              hipStream_t stream) {
    (void)in_sizes; (void)n_in; (void)out_size; (void)ws_size;

    const float* x     = (const float*)d_in[0];
    const float* Wqkv  = (const float*)d_in[1];
    const float* bqkv  = (const float*)d_in[2];
    const float* Wproj = (const float*)d_in[3];
    const float* bproj = (const float*)d_in[4];
    float* out = (float*)d_out;

    // workspace (72 MB total, all bf16):
    //   [0,16MB)   xb  (x as bf16)  -- reused as y_att after attention
    //   [16,22MB)  Wqkv bf16
    //   [22,24MB)  Wproj bf16
    //   [24,40MB)  q  [B,H,T,D]
    //   [40,56MB)  k  [B,H,T,D]
    //   [56,72MB)  v^T [B,H,D,T]
    char* ws = (char*)d_ws;
    __bf16* xb  = (__bf16*)(ws);
    __bf16* wqb = (__bf16*)(ws + ((size_t)16 << 20));
    __bf16* wpb = (__bf16*)(ws + ((size_t)22 << 20));
    __bf16* qb  = (__bf16*)(ws + ((size_t)24 << 20));
    __bf16* kb  = (__bf16*)(ws + ((size_t)40 << 20));
    __bf16* vtb = (__bf16*)(ws + ((size_t)56 << 20));
    __bf16* ya  = xb;   // alias: x consumed before attention writes

    dim3 blk(256);
    // upfront fp32 -> bf16 conversions (bandwidth-trivial)
    f32_to_bf16<<<dim3((8192 * 1024 / 4) / 256), blk, 0, stream>>>(x, xb, 8192 * 1024 / 4);
    f32_to_bf16<<<dim3((3072 * 1024 / 4) / 256), blk, 0, stream>>>(Wqkv, wqb, 3072 * 1024 / 4);
    f32_to_bf16<<<dim3((1024 * 1024 / 4) / 256), blk, 0, stream>>>(Wproj, wpb, 1024 * 1024 / 4);
    // QKV projection: M=8192, N=3072, K=1024
    gemm_wmma<0><<<dim3(24, 64), blk, 0, stream>>>(
        xb, wqb, bqkv, qb, kb, vtb, nullptr, 3072, 1024);
    // attention: grid (T/128, H, B)
    flash_attn<<<dim3(16, 16, 4), blk, 0, stream>>>(qb, kb, vtb, ya);
    // output projection: M=8192, N=1024, K=1024
    gemm_wmma<1><<<dim3(8, 64), blk, 0, stream>>>(
        ya, wpb, bproj, nullptr, nullptr, nullptr, out, 1024, 1024);
}
